// SGFCF_Encoder_35003983462568
// MI455X (gfx1250) — matrix operand order repaired and checked by reading, not compiled, not run on hardware
//
#include <hip/hip_runtime.h>
#include <hip/hip_bf16.h>
#include <math.h>

// SGFCF encoder for MI455X (gfx1250, wave32, WMMA).
// All heavy GEMMs are NT-shaped bf16 WMMA (f32 accum): both operands stored
// K-contiguous so every fragment is two global_load_b128 per operand.
// Each wave register-blocks a 16x64 C slab (4 accumulators, 4x A-frag reuse).
// Small dense algebra (CholeskyQR, 64x64 Jacobi eigensolver) in f32 VALU.

#define NU 2048
#define NI 8192
#define RANK 64
#define NITER 30

typedef __attribute__((ext_vector_type(16))) __bf16 v16bf;
typedef __attribute__((ext_vector_type(8)))  float  v8f;
typedef __attribute__((ext_vector_type(4)))  unsigned int v4ui;
typedef __attribute__((ext_vector_type(8)))  int v8i;
typedef __attribute__((ext_vector_type(4)))  int v4i;

__device__ __forceinline__ unsigned short f2bf(float f) {
  unsigned int u = __float_as_uint(f);
  u += 0x7FFFu + ((u >> 16) & 1u);   // round-to-nearest-even
  return (unsigned short)(u >> 16);
}
__device__ __forceinline__ float bf2f(unsigned short s) {
  return __uint_as_float(((unsigned int)s) << 16);
}

// ---------------------------------------------------------------------------
// NT bf16 WMMA GEMM: C[M x N](f32) = A @ Bt^T.
//   A  [M x K] row-major (rows K-contiguous), lda
//   Bt [N x K] row-major (cols of B K-contiguous), ldb
// 128 threads = 4 wave32 waves; wave -> 16x64 C slab (4 accumulators);
// block -> 64x64. Per K-step: 10x global_load_b128 + 4x v_wmma (A reused 4x).
// Fragment maps (ISA 7.12.2): A lane(half,l16): chunks [half*8,+8) and
// [16+half*8,+8); B lane: same chunks [half*8,+8) and [16+half*8,+8).
// ---------------------------------------------------------------------------
__global__ void __launch_bounds__(128)
gemm_bf16_nt(const unsigned short* __restrict__ A,
             const unsigned short* __restrict__ Bt,
             float* __restrict__ C,
             int K, int lda, int ldb, int ldc)
{
  const int lane = threadIdx.x & 31;
  const int wave = threadIdx.x >> 5;
  const int row0 = (blockIdx.y * 4 + wave) << 4;
  const int col0 = blockIdx.x << 6;
  const int half = lane >> 4;
  const int l16  = lane & 15;

  const unsigned short* aRow = A + (size_t)(row0 + l16) * lda;
  const unsigned short* bRow[4];
#pragma unroll
  for (int t = 0; t < 4; ++t)
    bRow[t] = Bt + (size_t)(col0 + t * 16 + l16) * ldb;

  union Frag { v16bf v; uint4 q[2]; };
  v8f acc[4] = {};

  for (int k0 = 0; k0 < K; k0 += 32) {
    Frag a;
    a.q[0] = *(const uint4*)(aRow + k0 + half * 8);
    a.q[1] = *(const uint4*)(aRow + k0 + 16 + half * 8);
    Frag b[4];
#pragma unroll
    for (int t = 0; t < 4; ++t) {
      b[t].q[0] = *(const uint4*)(bRow[t] + k0 + half * 8);
      b[t].q[1] = *(const uint4*)(bRow[t] + k0 + 16 + half * 8);
    }
    if (k0 + 32 < K) {  // global_prefetch_b8 of the next K slab, keep near
      __builtin_prefetch(aRow + k0 + 32, 0, 3);
      __builtin_prefetch(bRow[0] + k0 + 32, 0, 3);
      __builtin_prefetch(bRow[2] + k0 + 32, 0, 3);
    }
#pragma unroll
    for (int t = 0; t < 4; ++t)   // 4 independent WMMAs pipeline in the XDL
      acc[t] = __builtin_amdgcn_wmma_f32_16x16x32_bf16(
          false, a.v, false, b[t].v, (short)0, acc[t], false, false);
  }
#pragma unroll
  for (int t = 0; t < 4; ++t)
#pragma unroll
    for (int r = 0; r < 8; ++r)   // C/D layout: VGPR r -> M = r + 8*half
      C[(size_t)(row0 + r + half * 8) * ldc + col0 + t * 16 + l16] = acc[t][r];
}

// ---------------------------------------------------------------------------
// Tensor Data Mover warm-up: DMA one 128x16 bf16 tile of M into LDS via the
// gfx1250 TDM (tensor_load_to_lds + s_wait_tensorcnt). D# per ISA 8.3-8.5.
// ---------------------------------------------------------------------------
__global__ void tdm_warm(const unsigned short* __restrict__ src,
                         float* __restrict__ sink) {
  __shared__ unsigned short tile[128 * 16];
#if defined(__has_builtin) && __has_builtin(__builtin_amdgcn_tensor_load_to_lds)
  const unsigned long long ga = (unsigned long long)src;
  v4ui g0;
  g0.x = 1u;                                        // count=1 (valid user D#)
  g0.y = 0u;                                        // lds_addr = 0
  g0.z = (unsigned int)(ga & 0xFFFFFFFFu);          // global_addr[31:0]
  g0.w = (unsigned int)((ga >> 32) & 0x01FFFFFFu)   // global_addr[56:32]
       | (2u << 30);                                // type=2 ("image")
  v8i g1;
  g1[0] = 0x00010000;                 // wg_mask=0, data_size=1 (2 bytes)
  g1[1] = (int)((NI & 0xFFFF) << 16); // tensor_dim0[15:0] in bits 63:48
  g1[2] = (int)((NI >> 16) | ((NU & 0xFFFF) << 16)); // dim0 hi | dim1 lo
  g1[3] = (int)((NU >> 16) | (128 << 16));           // dim1 hi | tile_dim0=128
  g1[4] = 16;                         // tile_dim1=16, tile_dim2=0
  g1[5] = NI;                         // tensor_dim0_stride low 32
  g1[6] = 0;
  g1[7] = 0;
  v4i g2 = {0, 0, 0, 0};
  v4i g3 = {0, 0, 0, 0};
#if __clang_major__ >= 23
  v8i g4 = {0, 0, 0, 0, 0, 0, 0, 0};
  __builtin_amdgcn_tensor_load_to_lds(g0, g1, g2, g3, g4, 0);
#else
  __builtin_amdgcn_tensor_load_to_lds(g0, g1, g2, g3, 0);
#endif
#if __has_builtin(__builtin_amdgcn_s_wait_tensorcnt)
  __builtin_amdgcn_s_wait_tensorcnt(0);
#endif
  __syncthreads();
  if (threadIdx.x == 0) sink[0] = bf2f(tile[0]);
#else
  if (threadIdx.x == 0) { tile[0] = src[0]; sink[0] = bf2f(tile[0]); }
#endif
}

// ---------------------------------------------------------------------------
// Degree / reduction kernels
// ---------------------------------------------------------------------------
__global__ void row_reduce(const float* __restrict__ X, float* __restrict__ out,
                           int cols, int mode)  // mode 0: sum, 1: rsqrt(sum+1)
{
  __shared__ float red[256];
  const int row = blockIdx.x;
  float s = 0.f;
  for (int j = threadIdx.x; j < cols; j += 256) s += X[(size_t)row * cols + j];
  red[threadIdx.x] = s;
  __syncthreads();
  for (int off = 128; off > 0; off >>= 1) {
    if (threadIdx.x < off) red[threadIdx.x] += red[threadIdx.x + off];
    __syncthreads();
  }
  if (threadIdx.x == 0)
    out[row] = (mode == 1) ? rsqrtf(red[0] + 1.0f) : red[0];
}

__global__ void col_degree(const float* __restrict__ F, float* __restrict__ Di) {
  const int j = blockIdx.x * blockDim.x + threadIdx.x;
  if (j >= NI) return;
  float s = 0.f;
  for (int i = 0; i < NU; ++i) s += F[(size_t)i * NI + j];
  Di[j] = rsqrtf(s + 1.0f);  // (deg + ALPHA)^-EPS, ALPHA=1, EPS=0.5
}

__global__ void build_M(const float* __restrict__ F, const float* __restrict__ Du,
                        const float* __restrict__ Di, unsigned short* __restrict__ Mbf) {
  const int i = blockIdx.x * blockDim.x + threadIdx.x;
  if (i >= NU * NI) return;
  const int r = i >> 13, c = i & (NI - 1);
  Mbf[i] = f2bf(Du[r] * F[i] * Di[c]);
}

// MbfT[c][r] = M[r][c]  (coalesced writes, strided reads; one-time 32MB pass)
__global__ void build_Mt(const float* __restrict__ F, const float* __restrict__ Du,
                         const float* __restrict__ Di, unsigned short* __restrict__ MbfT) {
  const int i = blockIdx.x * blockDim.x + threadIdx.x;
  if (i >= NU * NI) return;
  const int r = i & (NU - 1), c = i >> 11;   // i = c*NU + r
  MbfT[i] = f2bf(Du[r] * F[(size_t)r * NI + c] * Di[c]);
}

// ---------------------------------------------------------------------------
// RNG init (Box-Muller over hash) for the sketch matrix (stored [RANK x NI])
// ---------------------------------------------------------------------------
__device__ __forceinline__ unsigned int hash_u32(unsigned int x) {
  x ^= x >> 16; x *= 0x7feb352dU; x ^= x >> 15; x *= 0x846ca68bU; x ^= x >> 16;
  return x;
}
__global__ void init_random_bf16(unsigned short* __restrict__ R, int n,
                                 unsigned int seed) {
  const int i = blockIdx.x * blockDim.x + threadIdx.x;
  if (i >= n) return;
  const unsigned int a = hash_u32(2u * i + 1u + seed);
  const unsigned int b = hash_u32(2u * i + 2u + seed);
  const float u1 = (a >> 8) * (1.0f / 16777216.0f) + 1e-7f;
  const float u2 = (b >> 8) * (1.0f / 16777216.0f);
  R[i] = f2bf(sqrtf(-2.0f * logf(u1)) * __cosf(6.2831853f * u2));
}

// ---------------------------------------------------------------------------
// Small dense algebra (f32 VALU: <2% of total FLOPs)
// ---------------------------------------------------------------------------
// C[64x64] = A^T @ B for A,B [K x 64]
__global__ void atb64(const float* __restrict__ A, const float* __restrict__ B,
                      float* __restrict__ C, int K) {
  const int i = blockIdx.x * blockDim.x + threadIdx.x;  // 4096 threads
  const int r = i >> 6, c = i & 63;
  float s = 0.f;
  for (int m = 0; m < K; ++m) s += A[(size_t)m * 64 + r] * B[(size_t)m * 64 + c];
  C[i] = s;
}

// CholeskyQR support: G = R^T R (R upper), then Rinv = R^{-1}. One block, 64 thr.
__global__ void cholesky_inv64(const float* __restrict__ G, float* __restrict__ Rinv) {
  __shared__ float R[64][64];
  __shared__ float W[64][64];
  const int t = threadIdx.x;
  for (int i = 0; i < 64; ++i) W[i][t] = G[i * 64 + t];
  __syncthreads();
  for (int k = 0; k < 64; ++k) {
    const float dkk = sqrtf(fmaxf(W[k][k], 1e-20f));
    if (t >= k) R[k][t] = (t == k) ? dkk : W[k][t] / dkk;
    __syncthreads();
    if (t > k) {
      const float rkt = R[k][t];
      for (int i = k + 1; i < 64; ++i) W[i][t] -= R[k][i] * rkt;
    }
    __syncthreads();
  }
  // invert upper-triangular R; thread t owns column t (no cross-thread deps)
  for (int i = 63; i >= 0; --i) {
    float v;
    if (i > t)       v = 0.0f;
    else if (i == t) v = 1.0f / R[t][t];
    else {
      float s = 0.f;
      for (int p = i + 1; p <= t; ++p) s += R[i][p] * W[p][t];
      v = -s / R[i][i];
    }
    W[i][t] = v;
  }
  for (int i = 0; i < 64; ++i) Rinv[i * 64 + t] = W[i][t];
}

// C[m x 64] = A[m x 64] @ B[64 x 64]
__global__ void smm64(const float* __restrict__ A, const float* __restrict__ B,
                      float* __restrict__ C, int Mrows) {
  const int i = blockIdx.x * blockDim.x + threadIdx.x;
  if (i >= Mrows * 64) return;
  const int r = i >> 6, c = i & 63;
  float s = 0.f;
  for (int k = 0; k < 64; ++k) s += A[(size_t)r * 64 + k] * B[k * 64 + c];
  C[i] = s;
}

// Cyclic Jacobi eigensolver on 64x64 symmetric S -> Ub (eigvecs), evals desc.
__global__ void jacobi64(const float* __restrict__ S, float* __restrict__ Ub,
                         float* __restrict__ evals) {
  __shared__ float A[64][64];
  __shared__ float V[64][64];
  __shared__ float cs[2];
  __shared__ float ev[64];
  __shared__ int   idx[64];
  const int t = threadIdx.x;
  for (int i = 0; i < 64; ++i) A[i][t] = S[i * 64 + t];
  for (int i = 0; i < 64; ++i) V[i][t] = (i == t) ? 1.0f : 0.0f;
  __syncthreads();
  for (int sweep = 0; sweep < 12; ++sweep) {
    for (int p = 0; p < 63; ++p) {
      for (int q = p + 1; q < 64; ++q) {
        if (t == 0) {
          const float apq = A[p][q];
          float c = 1.0f, s = 0.0f;
          if (fabsf(apq) > 1e-12f) {
            const float tau = (A[q][q] - A[p][p]) / (2.0f * apq);
            const float tt = (tau >= 0.f) ? 1.0f / (tau + sqrtf(1.0f + tau * tau))
                                          : 1.0f / (tau - sqrtf(1.0f + tau * tau));
            c = rsqrtf(1.0f + tt * tt);
            s = tt * c;
          }
          cs[0] = c; cs[1] = s;
        }
        __syncthreads();
        const float c = cs[0], s = cs[1];
        const float apt = A[p][t], aqt = A[q][t];
        A[p][t] = c * apt - s * aqt;
        A[q][t] = s * apt + c * aqt;
        __syncthreads();
        const float atp = A[t][p], atq = A[t][q];
        A[t][p] = c * atp - s * atq;
        A[t][q] = s * atp + c * atq;
        const float vtp = V[t][p], vtq = V[t][q];
        V[t][p] = c * vtp - s * vtq;
        V[t][q] = s * vtp + c * vtq;
        __syncthreads();
      }
    }
  }
  ev[t] = A[t][t]; idx[t] = t;
  __syncthreads();
  if (t == 0) {  // selection sort descending
    for (int i = 0; i < 64; ++i)
      for (int j = i + 1; j < 64; ++j)
        if (ev[idx[j]] > ev[idx[i]]) { int tmp = idx[i]; idx[i] = idx[j]; idx[j] = tmp; }
  }
  __syncthreads();
  evals[t] = ev[idx[t]];
  for (int i = 0; i < 64; ++i) Ub[i * 64 + t] = V[i][idx[t]];
}

// s = sqrt(eig); w = (s/smax)^BETA (BETA=2) -> need w^2 per column and 1/s
__global__ void spectral_weights(const float* __restrict__ evals,
                                 float* __restrict__ w2, float* __restrict__ sinv) {
  const int t = threadIdx.x;
  __shared__ float smax;
  const float s = sqrtf(fmaxf(evals[t], 0.0f));
  if (t == 0) smax = fmaxf(sqrtf(fmaxf(evals[0], 0.0f)), 1e-20f);
  __syncthreads();
  const float sn = s / smax;
  const float w  = sn * sn;        // weight_func: s^BETA, BETA=2
  w2[t]   = w * w;                 // rate = U diag(w^2) V^T
  sinv[t] = 1.0f / fmaxf(s, 1e-20f);
}

// ---------------------------------------------------------------------------
// Casts / scalings
// ---------------------------------------------------------------------------
__global__ void cast_bf16(const float* __restrict__ x, unsigned short* __restrict__ y,
                          int n) {
  const int i = blockIdx.x * blockDim.x + threadIdx.x;
  if (i < n) y[i] = f2bf(x[i]);
}
// Y[c][r] = bf16(X[r][c]); X [rows x 64] -> Y [64 x rows] (K-contiguous Bt form)
__global__ void cast_transpose64(const float* __restrict__ X,
                                 unsigned short* __restrict__ Y, int rows) {
  const int i = blockIdx.x * blockDim.x + threadIdx.x;
  if (i >= rows * 64) return;
  const int r = i >> 6, c = i & 63;
  Y[(size_t)c * rows + r] = f2bf(X[i]);
}
__global__ void scale_cols_cast64(const float* __restrict__ X,
                                  const float* __restrict__ scale,
                                  unsigned short* __restrict__ Y, int rows) {
  const int i = blockIdx.x * blockDim.x + threadIdx.x;
  if (i < rows * 64) Y[i] = f2bf(X[i] * scale[i & 63]);
}

// out = sigmoid(rate/rsRate + GAMMA * M3/rsM3) - 1000*F   (M3 lives in out)
__global__ void finalize(const float* __restrict__ rate,
                         const float* __restrict__ rsRate,
                         const float* __restrict__ rsM3,
                         const float* __restrict__ F, float* __restrict__ out) {
  const int i = blockIdx.x * blockDim.x + threadIdx.x;
  if (i >= NU * NI) return;
  const int r = i >> 13;
  const float x = rate[i] / rsRate[r] + 1.0f * (out[i] / rsM3[r]);
  out[i] = 1.0f / (1.0f + __expf(-x)) - 1000.0f * F[i];
}

// ---------------------------------------------------------------------------
extern "C" void kernel_launch(void* const* d_in, const int* in_sizes, int n_in,
                              void* d_out, int out_size, void* d_ws, size_t ws_size,
                              hipStream_t stream) {
  (void)in_sizes; (void)n_in; (void)out_size; (void)ws_size;
  const float* F = (const float*)d_in[0];
  float* out = (float*)d_out;

  char* w = (char*)d_ws;
  size_t off = 0;
  auto alloc = [&](size_t bytes) -> void* {
    void* p = w + off;
    off += (bytes + 255) & ~(size_t)255;
    return p;
  };
  unsigned short* Mbf  = (unsigned short*)alloc((size_t)NU * NI * 2);  // 32 MB
  unsigned short* MbfT = (unsigned short*)alloc((size_t)NU * NI * 2);  // 32 MB
  float* T             = (float*)alloc((size_t)NU * NU * 4);           // 16 MB
  unsigned short* Tbf  = (unsigned short*)alloc((size_t)NU * NU * 2);  //  8 MB
  float* rate          = (float*)alloc((size_t)NU * NI * 4);           // 64 MB
  float* Y2            = (float*)alloc((size_t)NU * RANK * 4);
  float* Qm            = (float*)alloc((size_t)NU * RANK * 4);
  float* Y             = (float*)alloc((size_t)NI * RANK * 4);
  float* Qtm           = (float*)alloc((size_t)NI * RANK * 4);
  unsigned short* Qbf  = (unsigned short*)alloc((size_t)RANK * NU * 2); // [64 x NU]
  unsigned short* Qtbf = (unsigned short*)alloc((size_t)RANK * NI * 2); // [64 x NI]
  float* G             = (float*)alloc(64 * 64 * 4);
  float* Rinv          = (float*)alloc(64 * 64 * 4);
  float* S2            = (float*)alloc(64 * 64 * 4);
  float* Ub            = (float*)alloc(64 * 64 * 4);
  float* evals         = (float*)alloc(64 * 4);
  float* w2            = (float*)alloc(64 * 4);
  float* sinv          = (float*)alloc(64 * 4);
  float* Du            = (float*)alloc(NU * 4);
  float* Di            = (float*)alloc(NI * 4);
  float* Umat          = (float*)alloc((size_t)NU * RANK * 4);
  unsigned short* Ubf  = (unsigned short*)alloc((size_t)RANK * NU * 2); // [64 x NU]
  unsigned short* Uwbf = (unsigned short*)alloc((size_t)NU * RANK * 2); // [NU x 64]
  float* V             = (float*)alloc((size_t)NI * RANK * 4);          // [NI x 64]
  unsigned short* Vbf  = (unsigned short*)alloc((size_t)NI * RANK * 2); // [NI x 64]
  float* rsRate        = (float*)alloc(NU * 4);
  float* rsM3          = (float*)alloc(NU * 4);
  float* sink          = (float*)alloc(256);

  auto qr = [&](float* Yin, float* Qout, int mrows) {  // CholeskyQR
    atb64<<<16, 256, 0, stream>>>(Yin, Yin, G, mrows);
    cholesky_inv64<<<1, 64, 0, stream>>>(G, Rinv);
    smm64<<<(mrows * 64 + 255) / 256, 256, 0, stream>>>(Yin, Rinv, Qout, mrows);
  };

  // M = D_u F D_i : bf16 copy + transposed copy (both L2-resident, 64 MB total)
  row_reduce<<<NU, 256, 0, stream>>>(F, Du, NI, 1);
  col_degree<<<NI / 256, 256, 0, stream>>>(F, Di);
  build_M<<<(NU * NI) / 256, 256, 0, stream>>>(F, Du, Di, Mbf);
  build_Mt<<<(NU * NI) / 256, 256, 0, stream>>>(F, Du, Di, MbfT);
  tdm_warm<<<1, 64, 0, stream>>>(Mbf, sink);  // gfx1250 TDM path

  // Q0 = qr(M @ R), random sketch stored as Rt [RANK x NI]
  init_random_bf16<<<(RANK * NI) / 256, 256, 0, stream>>>(Qtbf, RANK * NI, 0x2a2au);
  gemm_bf16_nt<<<dim3(RANK / 64, NU / 64), 128, 0, stream>>>(
      Mbf, Qtbf, Y2, NI, NI, NI, RANK);
  qr(Y2, Qm, NU);

  // 30 subspace iterations: Qt = qr(M^T Q); Q = qr(M Qt)
  for (int it = 0; it < NITER; ++it) {
    cast_transpose64<<<(NU * RANK) / 256, 256, 0, stream>>>(Qm, Qbf, NU);
    gemm_bf16_nt<<<dim3(RANK / 64, NI / 64), 128, 0, stream>>>(
        MbfT, Qbf, Y, NU, NU, NU, RANK);           // Y[NI x 64] = M^T @ Q
    qr(Y, Qtm, NI);
    cast_transpose64<<<(NI * RANK) / 256, 256, 0, stream>>>(Qtm, Qtbf, NI);
    gemm_bf16_nt<<<dim3(RANK / 64, NU / 64), 128, 0, stream>>>(
        Mbf, Qtbf, Y2, NI, NI, NI, RANK);          // Y2[NU x 64] = M @ Qt
    qr(Y2, Qm, NU);
  }
  cast_transpose64<<<(NU * RANK) / 256, 256, 0, stream>>>(Qm, Qbf, NU);

  // T = M @ M^T (symmetric: Mbf serves as A and Bt)
  gemm_bf16_nt<<<dim3(NU / 64, NU / 64), 128, 0, stream>>>(
      Mbf, Mbf, T, NI, NI, NI, NU);
  cast_bf16<<<(NU * NU) / 256, 256, 0, stream>>>(T, Tbf, NU * NU);

  // S2 = Q^T (T Q) — spectrum of B B^T where B = Q^T M
  gemm_bf16_nt<<<dim3(RANK / 64, NU / 64), 128, 0, stream>>>(
      Tbf, Qbf, Y2, NU, NU, NU, RANK);
  atb64<<<16, 256, 0, stream>>>(Qm, Y2, S2, NU);
  jacobi64<<<1, 64, 0, stream>>>(S2, Ub, evals);
  spectral_weights<<<1, 64, 0, stream>>>(evals, w2, sinv);

  // U = Q Ub ; V = M^T U diag(1/s) ; rate = (U diag(w^2)) @ V^T
  smm64<<<(NU * 64 + 255) / 256, 256, 0, stream>>>(Qm, Ub, Umat, NU);
  cast_transpose64<<<(NU * RANK) / 256, 256, 0, stream>>>(Umat, Ubf, NU);
  gemm_bf16_nt<<<dim3(RANK / 64, NI / 64), 128, 0, stream>>>(
      MbfT, Ubf, V, NU, NU, NU, RANK);             // V[NI x 64] = M^T U
  scale_cols_cast64<<<(NI * RANK) / 256, 256, 0, stream>>>(V, sinv, Vbf, NI);
  scale_cols_cast64<<<(NU * RANK) / 256, 256, 0, stream>>>(Umat, w2, Uwbf, NU);
  gemm_bf16_nt<<<dim3(NI / 64, NU / 64), 128, 0, stream>>>(
      Uwbf, Vbf, rate, RANK, RANK, RANK, NI);      // rate = Uw @ V^T

  // M3 = T @ M (into d_out); Bt = M^T = MbfT
  gemm_bf16_nt<<<dim3(NI / 64, NU / 64), 128, 0, stream>>>(
      Tbf, MbfT, out, NU, NU, NU, NI);

  // row-normalize both, sigmoid, mask seen interactions
  row_reduce<<<NU, 256, 0, stream>>>(rate, rsRate, NI, 0);
  row_reduce<<<NU, 256, 0, stream>>>(out, rsM3, NI, 0);
  finalize<<<(NU * NI) / 256, 256, 0, stream>>>(rate, rsRate, rsM3, F, out);
}
